// Group_query_Attention_26895085208065
// MI455X (gfx1250) — matrix operand order repaired and checked by reading, not compiled, not run on hardware
//
#include <hip/hip_runtime.h>
#include <hip/hip_bf16.h>

typedef __attribute__((ext_vector_type(16))) __bf16 v16bf;
typedef __attribute__((ext_vector_type(8)))  float  v8f;
typedef __attribute__((ext_vector_type(4)))  int    v4i;

#define B_   2
#define T_   2048
#define E_   2048
#define HQ_  32
#define HKV_ 8
#define HD_  64
#define G_   4
#define KVW_ (2*HKV_*HD_)   /* 1024: [K | V] concatenated per row */

#define AS1 __attribute__((address_space(1)))
#define AS3 __attribute__((address_space(3)))

// ---------------- CDNA5 async global->LDS copy (16B per lane) ----------------

__device__ __forceinline__ void async_cp16(const __bf16* g, __bf16* l) {
#if __has_builtin(__builtin_amdgcn_global_load_async_to_lds_b128)
  __builtin_amdgcn_global_load_async_to_lds_b128((AS1 v4i*)g, (AS3 v4i*)l, 0, 0);
#else
  *(int4*)l = *(const int4*)g;   // sync fallback through VGPRs
#endif
}

__device__ __forceinline__ void wait_async0() {
#if __has_builtin(__builtin_amdgcn_s_wait_asynccnt)
  __builtin_amdgcn_s_wait_asynccnt(0);
#else
  asm volatile("s_wait_asynccnt 0" ::: "memory");
#endif
}

// ---------------- wave32 half-wave reductions ----------------

__device__ __forceinline__ float half16_max(float v) {
  v = fmaxf(v, __shfl_xor(v, 1, 32));
  v = fmaxf(v, __shfl_xor(v, 2, 32));
  v = fmaxf(v, __shfl_xor(v, 4, 32));
  v = fmaxf(v, __shfl_xor(v, 8, 32));
  return v;
}
__device__ __forceinline__ float half16_sum(float v) {
  v += __shfl_xor(v, 1, 32);
  v += __shfl_xor(v, 2, 32);
  v += __shfl_xor(v, 4, 32);
  v += __shfl_xor(v, 8, 32);
  return v;
}

// ---------------- f32 -> bf16 convert ----------------

__global__ void cvt_bf16(const float* __restrict__ in, __bf16* __restrict__ out, int n) {
  int i = blockIdx.x * blockDim.x + threadIdx.x;
  if (i < n) out[i] = (__bf16)in[i];
}

// ---------------- GEMM: C[M,N] = A[M,K] * B[N,K]^T ----------------
// 256 threads = 8 waves; block tile 128x128; K-slices of 32 double-buffered in
// LDS via async copies. Wave tile 32x64 (2 A-frags x 4 B-frags = 8 WMMA/slice).

__global__ void __launch_bounds__(256)
gemm_nt_bf16(const __bf16* __restrict__ A, const __bf16* __restrict__ Bm,
             float* __restrict__ Cf, __bf16* __restrict__ Ch,
             int M, int N, int K, float scale) {
  __shared__ __bf16 Al[2][128 * 32];
  __shared__ __bf16 Bl[2][128 * 32];

  const int tid  = threadIdx.x;
  const int lane = tid & 31;
  const int wave = tid >> 5;
  const int half = lane >> 4;
  const int l15  = lane & 15;
  const int mw   = wave & 3;          // 4 x 32-row subtiles
  const int nw   = wave >> 2;         // 2 x 64-col subtiles
  const int nt128 = N >> 7;
  const int mt = blockIdx.x / nt128;
  const int nt = blockIdx.x % nt128;

  const __bf16* Ag = A  + (size_t)mt * 128 * K;
  const __bf16* Bg = Bm + (size_t)nt * 128 * K;

  auto stage = [&](int buf, int k0) {
    #pragma unroll
    for (int c = tid; c < 512; c += 256) {      // 128 rows x 4 x 16B each
      const int row = c >> 2;
      const int seg = (c & 3) * 8;              // element offset within 32-wide row
      async_cp16(Ag + (size_t)row * K + k0 + seg, &Al[buf][row * 32 + seg]);
      async_cp16(Bg + (size_t)row * K + k0 + seg, &Bl[buf][row * 32 + seg]);
    }
  };

  v8f acc[2][4] = {};
  stage(0, 0);

  for (int kk = 0; kk < K; kk += 32) {
    const int cur = (kk >> 5) & 1;
    wait_async0();
    __syncthreads();                            // slice `cur` visible to all waves
    if (kk + 32 < K) stage(cur ^ 1, kk + 32);   // prefetch next slice

    v16bf af[2];
    #pragma unroll
    for (int s = 0; s < 2; s++) {
      const __bf16* ar = &Al[cur][(mw * 32 + s * 16 + l15) * 32 + half * 8];
      #pragma unroll
      for (int i = 0; i < 8; i++) { af[s][i] = ar[i]; af[s][8 + i] = ar[16 + i]; }
    }
    #pragma unroll
    for (int nb = 0; nb < 4; nb++) {
      const __bf16* br = &Bl[cur][(nw * 64 + nb * 16 + l15) * 32 + half * 8];
      v16bf bf;
      #pragma unroll
      for (int i = 0; i < 8; i++) { bf[i] = br[i]; bf[8 + i] = br[16 + i]; }
      #pragma unroll
      for (int s = 0; s < 2; s++)
        acc[s][nb] = __builtin_amdgcn_wmma_f32_16x16x32_bf16(false, af[s], false, bf,
                                                             (short)0, acc[s][nb], false, false);
    }
  }

  #pragma unroll
  for (int s = 0; s < 2; s++)
    #pragma unroll
    for (int nb = 0; nb < 4; nb++)
      #pragma unroll
      for (int r = 0; r < 8; r++) {
        const int m = mt * 128 + mw * 32 + s * 16 + r + 8 * half;
        const int c = nt * 128 + nw * 64 + nb * 16 + l15;
        const float v = acc[s][nb][r] * scale;
        if (Cf) Cf[(size_t)m * N + c] = v;
        else    Ch[(size_t)m * N + c] = (__bf16)v;
      }
}

// ---------------- causal GQA flash attention ----------------
// Block = 4 waves = the 4 query heads of one KV group, same (b,h,16-query tile).
// K/V 32x64 tiles staged once per block via async-to-LDS, double buffered.
// Q has 1/sqrt(HD) folded in. KV row layout: [K(512) | V(512)].

__global__ void __launch_bounds__(128)
gqa_attn(const __bf16* __restrict__ Q, const __bf16* __restrict__ KV,
         __bf16* __restrict__ O) {
  __shared__ __bf16 Kl[2][32 * 64];
  __shared__ __bf16 Vl[2][32 * 64];
  __shared__ __bf16 Pl[4][16 * 32];

  const int tid  = threadIdx.x;
  const int lane = tid & 31;
  const int g    = tid >> 5;              // query head within the KV group
  const int half = lane >> 4;
  const int l15  = lane & 15;
  const int nQt  = T_ / 16;

  const int bid = blockIdx.x;             // over B * HKV * nQt
  const int b   = bid / (HKV_ * nQt);
  const int r0_ = bid % (HKV_ * nQt);
  const int h   = r0_ / nQt;
  const int q0  = (r0_ % nQt) * 16;
  const int hq  = h * G_ + g;

  const __bf16* kbase = KV + (size_t)(b * T_) * KVW_ + h * HD_;
  const __bf16* vbase = kbase + HKV_ * HD_;

  auto stage = [&](int buf, int sb) {
    #pragma unroll
    for (int c = tid; c < 256; c += 128) {  // 32 rows x 8 x 16B each (K and V)
      const int row = c >> 3;
      const int seg = (c & 7) * 8;
      async_cp16(kbase + (size_t)(sb + row) * KVW_ + seg, &Kl[buf][row * 64 + seg]);
      async_cp16(vbase + (size_t)(sb + row) * KVW_ + seg, &Vl[buf][row * 64 + seg]);
    }
  };

  // Q fragments (16 rows x HD=64 => two 16x32 A-frags), ISA A-layout
  const __bf16* qrow = Q + (size_t)(b * T_ + q0 + l15) * E_ + hq * HD_;
  v16bf qa0, qa1;
  #pragma unroll
  for (int i = 0; i < 8; i++) {
    qa0[i]     = qrow[half * 8 + i];
    qa0[8 + i] = qrow[half * 8 + 16 + i];
    qa1[i]     = qrow[32 + half * 8 + i];
    qa1[8 + i] = qrow[32 + half * 8 + 16 + i];
  }

  v8f acc[4] = {};
  float rm[8], rl[8];
  #pragma unroll
  for (int r = 0; r < 8; r++) { rm[r] = -__builtin_inff(); rl[r] = 0.0f; }

  __bf16* pl = &Pl[g][0];
  stage(0, 0);

  for (int sb = 0; sb <= q0 + 15; sb += 32) {   // same bound for all 4 waves
    const int cur = (sb >> 5) & 1;
    wait_async0();
    __syncthreads();
    if (sb + 32 <= q0 + 15) stage(cur ^ 1, sb + 32);

    // ---- S = Q K^T : two 16x16 tiles ----
    v8f sc[2];
    #pragma unroll
    for (int j = 0; j < 2; j++) {
      const __bf16* kr = &Kl[cur][(j * 16 + l15) * 64];
      v16bf kb0, kb1;
      #pragma unroll
      for (int i = 0; i < 16; i++) {
        kb0[i] = kr[half * 8 + (i & 7) + (i >> 3) * 16];        // d = 0..31
        kb1[i] = kr[32 + half * 8 + (i & 7) + (i >> 3) * 16];   // d = 32..63
      }
      v8f s = {};
      s = __builtin_amdgcn_wmma_f32_16x16x32_bf16(false, qa0, false, kb0, (short)0, s, false, false);
      s = __builtin_amdgcn_wmma_f32_16x16x32_bf16(false, qa1, false, kb1, (short)0, s, false, false);
      sc[j] = s;
    }

    // ---- causal mask + online softmax ----
    float p0[8], p1[8], alpha[8];
    #pragma unroll
    for (int r = 0; r < 8; r++) {
      const int t  = q0 + r + 8 * half;
      float v0 = (sb + l15      <= t) ? sc[0][r] : -__builtin_inff();
      float v1 = (sb + 16 + l15 <= t) ? sc[1][r] : -__builtin_inff();
      const float mx = half16_max(fmaxf(v0, v1));
      const float nm = fmaxf(rm[r], mx);
      const float al = __expf(rm[r] - nm);
      p0[r] = __expf(v0 - nm);
      p1[r] = __expf(v1 - nm);
      const float rs = half16_sum(p0[r] + p1[r]);
      rl[r] = rl[r] * al + rs;
      rm[r] = nm;
      alpha[r] = al;
    }
    #pragma unroll
    for (int nb = 0; nb < 4; nb++)
      #pragma unroll
      for (int r = 0; r < 8; r++) acc[nb][r] *= alpha[r];

    // ---- stage P (16x32) per-wave in LDS, reload as A-fragment ----
    #pragma unroll
    for (int r = 0; r < 8; r++) {
      const int m = r + 8 * half;
      pl[m * 32 + l15]      = (__bf16)p0[r];
      pl[m * 32 + 16 + l15] = (__bf16)p1[r];
    }
    v16bf pa;
    #pragma unroll
    for (int i = 0; i < 8; i++) {
      pa[i]     = pl[l15 * 32 + half * 16 + i];
      pa[8 + i] = pl[l15 * 32 + half * 16 + 8 + i];
    }

    // ---- O += P V ----
    #pragma unroll
    for (int nb = 0; nb < 4; nb++) {
      v16bf vf;
      #pragma unroll
      for (int i = 0; i < 16; i++)
        vf[i] = Vl[cur][(half * 16 + i) * 64 + nb * 16 + l15];
      acc[nb] = __builtin_amdgcn_wmma_f32_16x16x32_bf16(false, pa, false, vf,
                                                        (short)0, acc[nb], false, false);
    }
  }

  // ---- normalize, store bf16 at (row, hq*HD + d) ----
  #pragma unroll
  for (int nb = 0; nb < 4; nb++)
    #pragma unroll
    for (int r = 0; r < 8; r++) {
      const int m   = r + 8 * half;
      const int row = b * T_ + q0 + m;
      const int col = hq * HD_ + nb * 16 + l15;
      O[(size_t)row * E_ + col] = (__bf16)(acc[nb][r] / rl[r]);
    }
}

// ---------------- host launch ----------------

extern "C" void kernel_launch(void* const* d_in, const int* in_sizes, int n_in,
                              void* d_out, int out_size, void* d_ws, size_t ws_size,
                              hipStream_t stream) {
  const float* x   = (const float*)d_in[0];
  const float* Wq  = (const float*)d_in[1];
  const float* Wkv = (const float*)d_in[2];
  const float* Wo  = (const float*)d_in[3];
  float* out = (float*)d_out;

  const size_t nX   = (size_t)B_ * T_ * E_;
  const size_t nWq  = (size_t)E_ * E_;
  const size_t nWkv = (size_t)KVW_ * E_;
  const size_t nWo  = (size_t)E_ * E_;

  char* ws = (char*)d_ws;
  size_t off = 0;
  auto take = [&](size_t elems) {
    char* p = ws + off;
    off += (elems * sizeof(__bf16) + 255) & ~(size_t)255;
    return (__bf16*)p;
  };
  __bf16* xh   = take(nX);
  __bf16* wqh  = take(nWq);
  __bf16* wkvh = take(nWkv);
  __bf16* woh  = take(nWo);
  __bf16* qh   = take(nX);                      // (B*T, E)
  __bf16* kvh  = take((size_t)B_ * T_ * KVW_);  // (B*T, 1024)
  __bf16* oh   = take(nX);                      // attention output

  const int CT = 256;
  cvt_bf16<<<(int)((nX  + CT - 1) / CT), CT, 0, stream>>>(x,   xh,   (int)nX);
  cvt_bf16<<<(int)((nWq + CT - 1) / CT), CT, 0, stream>>>(Wq,  wqh,  (int)nWq);
  cvt_bf16<<<(int)((nWkv+ CT - 1) / CT), CT, 0, stream>>>(Wkv, wkvh, (int)nWkv);
  cvt_bf16<<<(int)((nWo + CT - 1) / CT), CT, 0, stream>>>(Wo,  woh,  (int)nWo);

  const int M = B_ * T_;                 // 4096
  const float qscale = 0.125f;           // 1/sqrt(HD)

  // q = x @ Wq^T (scale folded), bf16 out
  gemm_nt_bf16<<<(M / 128) * (E_ / 128), 256, 0, stream>>>(xh, wqh, nullptr, qh,
                                                           M, E_, E_, qscale);
  // kv = x @ Wkv^T, bf16 out
  gemm_nt_bf16<<<(M / 128) * (KVW_ / 128), 256, 0, stream>>>(xh, wkvh, nullptr, kvh,
                                                             M, KVW_, E_, 1.0f);
  // attention: one block per (b, kv-head, 16-query tile)
  gqa_attn<<<B_ * HKV_ * (T_ / 16), 128, 0, stream>>>(qh, kvh, oh);

  // out = attn @ Wo^T, f32 out
  gemm_nt_bf16<<<(M / 128) * (E_ / 128), 256, 0, stream>>>(oh, woh, out, nullptr,
                                                           M, E_, E_, 1.0f);
}
